// RoNGAT_NA_81767587381923
// MI455X (gfx1250) — matrix operand order repaired and checked by reading, not compiled, not run on hardware
//
#include <hip/hip_runtime.h>

typedef _Float16 f16;
typedef __attribute__((ext_vector_type(8)))  _Float16 v8h;
typedef __attribute__((ext_vector_type(16))) _Float16 v16h;
typedef __attribute__((ext_vector_type(8)))  float    v8f;

#define NN 50000
#define EE 800000
#define DD 128
#define EDD 64

// ---------------------------------------------------------------------------
// fragment helpers (wave32 WMMA 16x16x32 f16 layouts, CDNA5 ISA)
// A: lane half kbase = 0/8; elems 0-7 = K kbase..+7, elems 8-15 = K kbase+16..+23
// B (transposed storage Wt[n][k]): same K order as A (kbase = 0/8; elems 0-7 = kbase..+7, 8-15 = kbase+16..+23)
// D: VGPR r -> row r + 8*(lane>=16), col lane&15
// ---------------------------------------------------------------------------

static __device__ __forceinline__ v16h a_f16(const f16* __restrict__ row, int kb) {
    v8h lo = *(const v8h*)(row + kb);
    v8h hi = *(const v8h*)(row + kb + 16);
    return __builtin_shufflevector(lo, hi, 0,1,2,3,4,5,6,7,8,9,10,11,12,13,14,15);
}

static __device__ __forceinline__ v8h cvt8s(const float* __restrict__ p, float s) {
    float4 x = *(const float4*)p;
    float4 y = *(const float4*)(p + 4);
    v8h r;
    r[0] = (f16)(x.x * s); r[1] = (f16)(x.y * s); r[2] = (f16)(x.z * s); r[3] = (f16)(x.w * s);
    r[4] = (f16)(y.x * s); r[5] = (f16)(y.y * s); r[6] = (f16)(y.z * s); r[7] = (f16)(y.w * s);
    return r;
}

static __device__ __forceinline__ v16h a_f32_s(const float* __restrict__ row, int kb, float s) {
    v8h lo = cvt8s(row + kb, s);
    v8h hi = cvt8s(row + kb + 16, s);
    return __builtin_shufflevector(lo, hi, 0,1,2,3,4,5,6,7,8,9,10,11,12,13,14,15);
}

// works for LDS or global pointers (inlined; addrspace inferred per call site)
static __device__ __forceinline__ v16h b_t(const f16* Wt, int stride, int lane, int n0, int kk) {
    int n  = lane & 15;
    int kh = (lane >> 4) << 3;
    const f16* p = Wt + (n0 * 16 + n) * stride + kk * 32 + kh;
    v8h lo = *(const v8h*)p;
    v8h hi = *(const v8h*)(p + 16);
    return __builtin_shufflevector(lo, hi, 0,1,2,3,4,5,6,7,8,9,10,11,12,13,14,15);
}

static __device__ __forceinline__ v16h a_stage(const f16* z, int stride, int lane, int kk) {
    int m  = lane & 15;
    int kb = ((lane >> 4) << 3) + kk * 32;
    const f16* p = z + m * stride + kb;
    v8h lo = *(const v8h*)p;
    v8h hi = *(const v8h*)(p + 16);
    return __builtin_shufflevector(lo, hi, 0,1,2,3,4,5,6,7,8,9,10,11,12,13,14,15);
}

static __device__ __forceinline__ v8f wmma16(v16h a, v16h b, v8f c) {
    return __builtin_amdgcn_wmma_f32_16x16x32_f16(false, a, false, b, (short)0, c, false, false);
}

static __device__ __forceinline__ void wave_lds_fence() {
    asm volatile("s_wait_dscnt 0" ::: "memory");
}

// ---------------------------------------------------------------------------
// prep kernels
// ---------------------------------------------------------------------------
__global__ void zero_kernel(float* __restrict__ p, long n) {
    long i = (long)blockIdx.x * blockDim.x + threadIdx.x;
    long stride = (long)gridDim.x * blockDim.x;
    for (; i < n; i += stride) p[i] = 0.0f;
}

__global__ void cvt_f16_kernel(const float* __restrict__ in, f16* __restrict__ out, long n) {
    long i = (long)blockIdx.x * blockDim.x + threadIdx.x;
    long stride = (long)gridDim.x * blockDim.x;
    for (; i < n; i += stride) out[i] = (f16)in[i];
}

// W [K][srcN] fp32 row-major -> Wt [outN][stride] f16 transposed, zero padded
__global__ void prep_wt_kernel(const float* __restrict__ W, f16* __restrict__ Wt,
                               int K, int srcN, int outN, int stride) {
    int total = outN * stride;
    for (int i = blockIdx.x * blockDim.x + threadIdx.x; i < total;
         i += gridDim.x * blockDim.x) {
        int n = i / stride, k = i - n * stride;
        f16 v = (f16)0.0f;
        if (k < K && n < srcN) v = (f16)W[k * srcN + n];
        Wt[i] = v;
    }
}

// ---------------------------------------------------------------------------
// msg kernel: m = [h[src] | ef] @ Wm + bm ; atomic scatter to summed[dst]
// h, ef are f16; Wmt_ws is pre-transposed f16 [128][200] copied to LDS.
// All data-dependent addressing uses unsigned 32-bit element offsets so the
// backend can use the SADDR+voffset global addressing mode.
// ---------------------------------------------------------------------------
__global__ __launch_bounds__(256) void msg_kernel(
    const f16* __restrict__ h, const f16* __restrict__ ef,
    const int* __restrict__ src, const int* __restrict__ dst,
    const f16* __restrict__ Wmt_ws, const float* __restrict__ bm,
    float* __restrict__ summed, float* __restrict__ deg)
{
    __shared__ alignas(16) f16 Wmt[128 * 200];
    {
        const v8h* s = (const v8h*)Wmt_ws;
        v8h* d = (v8h*)Wmt;
        for (int i = threadIdx.x; i < 128 * 200 / 8; i += blockDim.x) d[i] = s[i];
    }
    __syncthreads();

    const int lane = threadIdx.x & 31;
    const int m    = lane & 15;
    const int lh   = lane >> 4;
    const int kb   = lh << 3;
    const int wid    = blockIdx.x * (blockDim.x >> 5) + (threadIdx.x >> 5);
    const int nwaves = gridDim.x * (blockDim.x >> 5);

    float biasv[8];
    #pragma unroll
    for (int n0 = 0; n0 < 8; ++n0) biasv[n0] = bm[n0 * 16 + m];

    for (int tile = wid; tile < EE / 16; tile += nwaves) {
        const int base = tile * 16;
        __builtin_prefetch(src + base + nwaves * 16, 0, 1);
        __builtin_prefetch(dst + base + nwaves * 16, 0, 1);
        const unsigned s = (unsigned)src[base + m];
        const unsigned e = (unsigned)(base + m);

        v16h a[6];
        const f16* hrow = h + s * (unsigned)DD;
        a[0] = a_f16(hrow +  0, kb);
        a[1] = a_f16(hrow + 32, kb);
        a[2] = a_f16(hrow + 64, kb);
        a[3] = a_f16(hrow + 96, kb);
        const f16* erow = ef + e * (unsigned)EDD;
        a[4] = a_f16(erow +  0, kb);
        a[5] = a_f16(erow + 32, kb);

        unsigned doff[8];
        #pragma unroll
        for (int r = 0; r < 8; ++r)
            doff[r] = (unsigned)dst[base + lh * 8 + r] * (unsigned)DD + (unsigned)m;
        if (lane < 16) atomicAdd(deg + (unsigned)dst[base + lane], 1.0f);

        #pragma unroll
        for (int n0 = 0; n0 < 8; ++n0) {
            v8f c;
            #pragma unroll
            for (int r = 0; r < 8; ++r) c[r] = biasv[n0];
            #pragma unroll
            for (int kk = 0; kk < 6; ++kk)
                c = wmma16(a[kk], b_t(Wmt, 200, lane, n0, kk), c);
            #pragma unroll
            for (int r = 0; r < 8; ++r)
                atomicAdd(summed + (doff[r] + (unsigned)(n0 * 16)), c[r]);
        }
    }
}

// ---------------------------------------------------------------------------
// apply kernel: h' = leaky_relu([h | summed/max(deg,1)] @ Wa + ba) -> f16
// wave w owns N-tile n0=w; 8 B-frags of pre-transposed Wa live in registers.
// ---------------------------------------------------------------------------
__global__ __launch_bounds__(256) void apply_kernel(
    const f16* __restrict__ h, const float* __restrict__ summed,
    const float* __restrict__ deg,
    const f16* __restrict__ Wat, const float* __restrict__ ba,
    f16* __restrict__ hout)
{
    const int lane = threadIdx.x & 31;
    const int n0   = threadIdx.x >> 5;
    const int m    = lane & 15;
    const int lh   = lane >> 4;
    const int kb   = lh << 3;
    const int col  = n0 * 16 + m;

    v16h b[8];
    #pragma unroll
    for (int kk = 0; kk < 8; ++kk) b[kk] = b_t(Wat, 264, lane, n0, kk);
    const float bias = ba[col];

    for (int tile = blockIdx.x; tile < NN / 16; tile += gridDim.x) {
        const unsigned v = (unsigned)(tile * 16 + m);
        float rd = 1.0f / fmaxf(deg[v], 1.0f);
        const f16*   hrow = h      + v * (unsigned)DD;
        const float* srow = summed + v * (unsigned)DD;

        v16h a[8];
        #pragma unroll
        for (int kk = 0; kk < 4; ++kk) a[kk]     = a_f16  (hrow + 32 * kk, kb);
        #pragma unroll
        for (int kk = 0; kk < 4; ++kk) a[kk + 4] = a_f32_s(srow + 32 * kk, kb, rd);

        v8f c;
        #pragma unroll
        for (int r = 0; r < 8; ++r) c[r] = bias;
        #pragma unroll
        for (int kk = 0; kk < 8; ++kk) c = wmma16(a[kk], b[kk], c);

        #pragma unroll
        for (int r = 0; r < 8; ++r) {
            float x = c[r];
            x = x > 0.0f ? x : 0.01f * x;
            unsigned row = (unsigned)(tile * 16 + r + lh * 8);
            hout[row * (unsigned)DD + (unsigned)col] = (f16)x;
        }
    }
}

// ---------------------------------------------------------------------------
// heads kernel: pair = [h[src] | h[dst]] (f16)
//   coarse = relu(pair@Wc1+bc1)@Wc2+bc2   (2nd layer WMMA, N padded 2->16)
//   fine   = relu(BN(pair@Wf1+bf1))@Wf2+bf2 (2nd layer WMMA, N padded 10->16)
// ---------------------------------------------------------------------------
__global__ __launch_bounds__(128) void heads_kernel(
    const f16* __restrict__ h, const int* __restrict__ src, const int* __restrict__ dst,
    const f16* __restrict__ Wc1t_ws, const float* __restrict__ bc1,
    const f16* __restrict__ Wc2t_ws, const float* __restrict__ bc2,
    const f16* __restrict__ Wf1t_ws, const float* __restrict__ bf1,
    const float* __restrict__ bng, const float* __restrict__ bnb,
    const float* __restrict__ bnm, const float* __restrict__ bnv,
    const f16* __restrict__ Wf2t_ws, const float* __restrict__ bf2,
    float* __restrict__ outc, float* __restrict__ outf)
{
    __shared__ alignas(16) f16 Wc1t[32 * 264];
    __shared__ alignas(16) f16 Wf1t[64 * 264];
    __shared__ alignas(16) f16 Wc2t[16 * 32];
    __shared__ alignas(16) f16 Wf2t[16 * 64];
    __shared__ alignas(16) f16 zst[4][16 * 72];

    for (int i = threadIdx.x; i < 32 * 264 / 8; i += blockDim.x)
        ((v8h*)Wc1t)[i] = ((const v8h*)Wc1t_ws)[i];
    for (int i = threadIdx.x; i < 64 * 264 / 8; i += blockDim.x)
        ((v8h*)Wf1t)[i] = ((const v8h*)Wf1t_ws)[i];
    for (int i = threadIdx.x; i < 16 * 32 / 8; i += blockDim.x)
        ((v8h*)Wc2t)[i] = ((const v8h*)Wc2t_ws)[i];
    for (int i = threadIdx.x; i < 16 * 64 / 8; i += blockDim.x)
        ((v8h*)Wf2t)[i] = ((const v8h*)Wf2t_ws)[i];
    __syncthreads();

    const int lane = threadIdx.x & 31;
    const int m    = lane & 15;
    const int lh   = lane >> 4;
    const int kb   = lh << 3;
    f16* z = &zst[threadIdx.x >> 5][0];

    const int wid    = blockIdx.x * (blockDim.x >> 5) + (threadIdx.x >> 5);
    const int nwaves = gridDim.x * (blockDim.x >> 5);

    // loop-invariant per-column scalars
    float biasc[2], scl[4], shf[4], biasf[4];
    #pragma unroll
    for (int n0 = 0; n0 < 2; ++n0) biasc[n0] = bc1[n0 * 16 + m];
    #pragma unroll
    for (int n0 = 0; n0 < 4; ++n0) {
        int k = n0 * 16 + m;
        scl[n0]   = bng[k] * rsqrtf(bnv[k] + 1e-5f);
        shf[n0]   = bnb[k] - bnm[k] * scl[n0];
        biasf[n0] = bf1[k];
    }
    const float bias2c = (m < 2)  ? bc2[m] : 0.0f;
    const float bias2f = (m < 10) ? bf2[m] : 0.0f;

    for (int tile = wid; tile < EE / 16; tile += nwaves) {
        const int base = tile * 16;
        __builtin_prefetch(src + base + nwaves * 16, 0, 1);
        __builtin_prefetch(dst + base + nwaves * 16, 0, 1);
        const unsigned es = (unsigned)src[base + m];
        const unsigned ed = (unsigned)dst[base + m];

        v16h a[8];
        const f16* sr = h + es * (unsigned)DD;
        const f16* dr = h + ed * (unsigned)DD;
        #pragma unroll
        for (int kk = 0; kk < 4; ++kk) a[kk]     = a_f16(sr + 32 * kk, kb);
        #pragma unroll
        for (int kk = 0; kk < 4; ++kk) a[kk + 4] = a_f16(dr + 32 * kk, kb);

        // ---- coarse layer 1: 256 -> 32, relu, stage ------------------------
        #pragma unroll
        for (int n0 = 0; n0 < 2; ++n0) {
            v8f c;
            #pragma unroll
            for (int r = 0; r < 8; ++r) c[r] = biasc[n0];
            #pragma unroll
            for (int kk = 0; kk < 8; ++kk)
                c = wmma16(a[kk], b_t(Wc1t, 264, lane, n0, kk), c);
            #pragma unroll
            for (int r = 0; r < 8; ++r)
                z[(r + lh * 8) * 72 + n0 * 16 + m] = (f16)fmaxf(c[r], 0.0f);
        }
        wave_lds_fence();

        // ---- coarse layer 2: 32 -> 2 (padded 16), one WMMA -----------------
        {
            v16h az = a_stage(z, 72, lane, 0);
            v16h bz = b_t(Wc2t, 32, lane, 0, 0);
            v8f c;
            #pragma unroll
            for (int r = 0; r < 8; ++r) c[r] = bias2c;
            c = wmma16(az, bz, c);
            if (m < 2) {
                #pragma unroll
                for (int r = 0; r < 8; ++r) {
                    unsigned eo = (unsigned)(base + r + lh * 8) * 2u + (unsigned)m;
                    outc[eo] = c[r];
                }
            }
        }
        wave_lds_fence();

        // ---- fine layer 1: 256 -> 64, BN + relu, stage ---------------------
        #pragma unroll
        for (int n0 = 0; n0 < 4; ++n0) {
            v8f c;
            #pragma unroll
            for (int r = 0; r < 8; ++r) c[r] = biasf[n0];
            #pragma unroll
            for (int kk = 0; kk < 8; ++kk)
                c = wmma16(a[kk], b_t(Wf1t, 264, lane, n0, kk), c);
            #pragma unroll
            for (int r = 0; r < 8; ++r) {
                float x = c[r] * scl[n0] + shf[n0];
                z[(r + lh * 8) * 72 + n0 * 16 + m] = (f16)fmaxf(x, 0.0f);
            }
        }
        wave_lds_fence();

        // ---- fine layer 2: 64 -> 10 (padded 16), two WMMAs -----------------
        {
            v8f c;
            #pragma unroll
            for (int r = 0; r < 8; ++r) c[r] = bias2f;
            #pragma unroll
            for (int kk = 0; kk < 2; ++kk)
                c = wmma16(a_stage(z, 72, lane, kk), b_t(Wf2t, 64, lane, 0, kk), c);
            if (m < 10) {
                #pragma unroll
                for (int r = 0; r < 8; ++r) {
                    unsigned eo = (unsigned)(base + r + lh * 8) * 10u + (unsigned)m;
                    outf[eo] = c[r];
                }
            }
        }
        wave_lds_fence();
    }
}

// ---------------------------------------------------------------------------
// launcher
// ---------------------------------------------------------------------------
extern "C" void kernel_launch(void* const* d_in, const int* in_sizes, int n_in,
                              void* d_out, int out_size, void* d_ws, size_t ws_size,
                              hipStream_t stream) {
    (void)in_sizes; (void)n_in; (void)out_size; (void)ws_size;

    const float* nfeats = (const float*)d_in[0];
    const float* efeats = (const float*)d_in[1];
    const int*   src    = (const int*)d_in[2];
    const int*   dst    = (const int*)d_in[3];
    const float* Wm1 = (const float*)d_in[4];  const float* bm1 = (const float*)d_in[5];
    const float* Wa1 = (const float*)d_in[6];  const float* ba1 = (const float*)d_in[7];
    const float* Wm2 = (const float*)d_in[8];  const float* bm2 = (const float*)d_in[9];
    const float* Wa2 = (const float*)d_in[10]; const float* ba2 = (const float*)d_in[11];
    const float* Wc1 = (const float*)d_in[12]; const float* bc1 = (const float*)d_in[13];
    const float* Wc2 = (const float*)d_in[14]; const float* bc2 = (const float*)d_in[15];
    const float* Wf1 = (const float*)d_in[16]; const float* bf1 = (const float*)d_in[17];
    const float* bng = (const float*)d_in[18]; const float* bnb = (const float*)d_in[19];
    const float* bnm = (const float*)d_in[20]; const float* bnv = (const float*)d_in[21];
    const float* Wf2 = (const float*)d_in[22]; const float* bf2 = (const float*)d_in[23];

    // workspace layout (all 16B aligned)
    char* w = (char*)d_ws;
    float* summed = (float*)w;  w += (size_t)NN * DD * 4;
    float* deg    = (float*)w;  w += (size_t)NN * 4;
    f16* h1   = (f16*)w;        w += (size_t)NN * DD * 2;
    f16* h2   = (f16*)w;        w += (size_t)NN * DD * 2;
    f16* nf16 = (f16*)w;        w += (size_t)NN * DD * 2;
    f16* ef16 = (f16*)w;        w += (size_t)EE * EDD * 2;
    f16* Wm1t = (f16*)w;        w += 128 * 200 * 2;
    f16* Wm2t = (f16*)w;        w += 128 * 200 * 2;
    f16* Wa1t = (f16*)w;        w += 128 * 264 * 2;
    f16* Wa2t = (f16*)w;        w += 128 * 264 * 2;
    f16* Wc1t = (f16*)w;        w += 32 * 264 * 2;
    f16* Wf1t = (f16*)w;        w += 64 * 264 * 2;
    f16* Wc2t = (f16*)w;        w += 16 * 32 * 2;
    f16* Wf2t = (f16*)w;        w += 16 * 64 * 2;

    float* outc = (float*)d_out;            // [E,2]
    float* outf = outc + (size_t)2 * EE;    // [E,10]

    // ---- prep: f16 features + transposed f16 weights -----------------------
    cvt_f16_kernel<<<1024, 256, 0, stream>>>(nfeats, nf16, (long)NN * DD);
    cvt_f16_kernel<<<2048, 256, 0, stream>>>(efeats, ef16, (long)EE * EDD);
    prep_wt_kernel<<<128, 256, 0, stream>>>(Wm1, Wm1t, 192, 128, 128, 200);
    prep_wt_kernel<<<128, 256, 0, stream>>>(Wm2, Wm2t, 192, 128, 128, 200);
    prep_wt_kernel<<<128, 256, 0, stream>>>(Wa1, Wa1t, 256, 128, 128, 264);
    prep_wt_kernel<<<128, 256, 0, stream>>>(Wa2, Wa2t, 256, 128, 128, 264);
    prep_wt_kernel<<<64,  256, 0, stream>>>(Wc1, Wc1t, 256, 32, 32, 264);
    prep_wt_kernel<<<64,  256, 0, stream>>>(Wf1, Wf1t, 256, 64, 64, 264);
    prep_wt_kernel<<<8,   256, 0, stream>>>(Wc2, Wc2t, 32, 2, 16, 32);
    prep_wt_kernel<<<8,   256, 0, stream>>>(Wf2, Wf2t, 64, 10, 16, 64);

    const long zn = (long)NN * DD + NN;

    // ---- layer 1 -----------------------------------------------------------
    zero_kernel<<<2048, 256, 0, stream>>>(summed, zn);
    msg_kernel<<<512, 256, 0, stream>>>(nf16, ef16, src, dst, Wm1t, bm1, summed, deg);
    apply_kernel<<<512, 256, 0, stream>>>(nf16, summed, deg, Wa1t, ba1, h1);
    // ---- layer 2 -----------------------------------------------------------
    zero_kernel<<<2048, 256, 0, stream>>>(summed, zn);
    msg_kernel<<<512, 256, 0, stream>>>(h1, ef16, src, dst, Wm2t, bm2, summed, deg);
    apply_kernel<<<512, 256, 0, stream>>>(h1, summed, deg, Wa2t, ba2, h2);
    // ---- edge heads --------------------------------------------------------
    heads_kernel<<<1024, 128, 0, stream>>>(h2, src, dst,
                                           Wc1t, bc1, Wc2t, bc2,
                                           Wf1t, bf1, bng, bnb, bnm, bnv, Wf2t, bf2,
                                           outc, outf);
}
